// AxisAttention_73332271612216
// MI455X (gfx1250) — matrix operand order repaired and checked by reading, not compiled, not run on hardware
//
#include <hip/hip_runtime.h>
#include <hip/hip_bf16.h>

typedef __attribute__((ext_vector_type(16))) _Float16 v16h;
typedef __attribute__((ext_vector_type(8)))  float    v8f;

union Frag16 { v16h v; unsigned u[8]; };

// ---------------------------------------------------------------------------
// CDNA5 async global->LDS copy (per-lane 16B), tracked by ASYNCcnt.
// lds_byte_addr: LDS byte offset in a VGPR; gaddr: 64-bit global address.
// ---------------------------------------------------------------------------
__device__ __forceinline__ void async_copy_b128(unsigned lds_byte_addr,
                                                const void* gaddr) {
    asm volatile("global_load_async_to_lds_b128 %0, %1, off"
                 :: "v"(lds_byte_addr), "v"((unsigned long long)gaddr)
                 : "memory");
}
__device__ __forceinline__ void wait_async0() {
    asm volatile("s_wait_asynccnt 0x0" ::: "memory");
}
__device__ __forceinline__ unsigned lds_addr_of(const void* p) {
    return (unsigned)(unsigned long long)p;   // low 32 bits = LDS byte offset
}

// ---------------------------------------------------------------------------
// Kernel 1: QKV projection.  out[o,s] = sum_c W[o,c] * x[b,c,s] + bias[o]
// M = o (192 -> 12 tiles), K = c (64 -> 2 steps of 32), N = s (128/WG).
// Writes qT,kT as [bnd][w][h] (transposed) and vM as [bnd][h][w], all f16.
// ---------------------------------------------------------------------------
__global__ __launch_bounds__(256) void qkv_proj_kernel(
    const float* __restrict__ x, const float* __restrict__ Wq,
    const float* __restrict__ bq,
    _Float16* __restrict__ qT, _Float16* __restrict__ kT,
    _Float16* __restrict__ vM)
{
    __shared__ __align__(16) _Float16 Wh[192 * 64];   // [o][c]  24 KB
    __shared__ __align__(16) _Float16 Xh[128 * 64];   // [s][c]  16 KB
    __shared__ float Bh[192];

    const int tid   = threadIdx.x;
    const int b     = blockIdx.y;
    const int sBase = blockIdx.x * 128;

    for (int i = tid; i < 192 * 64; i += 256) Wh[i] = (_Float16)Wq[i];
    if (tid < 192) Bh[tid] = bq[tid];
    for (int i = tid; i < 128 * 64; i += 256) {
        int c = i >> 7, s = i & 127;
        Xh[s * 64 + c] = (_Float16)x[(size_t)(b * 64 + c) * 65536 + sBase + s];
    }
    __syncthreads();

    const int lane = tid & 31, wv = tid >> 5;       // 8 waves
    const int m = lane & 15, half = lane >> 4, nn = m;
    const int s = sBase + wv * 16 + nn;
    const int hrow = s >> 8, wcol = s & 255;

    for (int ot = 0; ot < 12; ++ot) {
        v8f acc = {};
        #pragma unroll
        for (int ks = 0; ks < 2; ++ks) {
            Frag16 A, Bf;
            #pragma unroll
            for (int v = 0; v < 8; ++v) {
                int ka = ks * 32 + ((v & 3) * 2) + ((v >> 2) * 16) + half * 8;
                A.u[v]  = *(const unsigned*)&Wh[(ot * 16 + m) * 64 + ka];
                int kb = ks * 32 + ((v & 3) * 2) + ((v >> 2) * 16) + half * 8;
                Bf.u[v] = *(const unsigned*)&Xh[(wv * 16 + nn) * 64 + kb];
            }
            acc = __builtin_amdgcn_wmma_f32_16x16x32_f16(
                false, A.v, false, Bf.v, (short)0, acc, false, false);
        }
        #pragma unroll
        for (int r = 0; r < 8; ++r) {
            int o = ot * 16 + r + half * 8;
            float val = acc[r] + Bh[o];
            int cc  = o & 63;
            int bnd = (b * 4 + (cc >> 4)) * 16 + (cc & 15);
            size_t base = (size_t)bnd * 65536;
            if (o < 64)       qT[base + (size_t)wcol * 256 + hrow] = (_Float16)val;
            else if (o < 128) kT[base + (size_t)wcol * 256 + hrow] = (_Float16)val;
            else              vM[base + (size_t)hrow * 256 + wcol] = (_Float16)val;
        }
    }
}

// ---------------------------------------------------------------------------
// Kernel 2: S = Q^T K per (b,n,d) slice, softmax((S+pos)*0.25) row-wise,
// writes P^T [j][i] in f16 via LDS transpose + coalesced b128 stores.
// One WG (16 waves) per slice; K slice async-loaded into LDS.
// ---------------------------------------------------------------------------
__global__ __launch_bounds__(512) void attn_scores_kernel(
    const _Float16* __restrict__ qT, const _Float16* __restrict__ kT,
    const float* __restrict__ pos, _Float16* __restrict__ Pt)
{
    __shared__ __align__(16) _Float16 Kh[65536];  // K^T slice [j][h]  128 KB
    __shared__ __align__(16) _Float16 Qi[4096];   // Q rows / P^T xpose  8 KB
    __shared__ float Srow[4096];                  // S block [i][j]    16 KB

    const int tid = threadIdx.x;
    const int bnd = blockIdx.x;
    const int nd  = bnd & 63;                     // n*16 + d
    const size_t sliceBase = (size_t)bnd * 65536;

    // async 128-bit copies: global K^T slice -> LDS (16 per thread)
    {
        const unsigned khBase = lds_addr_of(Kh);
        const _Float16* src = kT + sliceBase;
        for (int i = tid; i < 8192; i += 512)
            async_copy_b128(khBase + i * 16, src + (size_t)i * 8);
    }

    const int lane = tid & 31, wv = tid >> 5;     // 16 waves, wave = j-tile
    const int m = lane & 15, half = lane >> 4, nn = m;
    const unsigned qiBase = lds_addr_of(Qi);
    _Float16* Tb = Qi;                            // aliased transpose buffer

    for (int ib = 0; ib < 16; ++ib) {
        __syncthreads();
        // stage 16 rows of Q^T (contiguous 8 KB), one async b128 per thread
        async_copy_b128(qiBase + tid * 16,
                        qT + sliceBase + (size_t)ib * 4096 + tid * 8);
        wait_async0();
        __syncthreads();

        v8f acc = {};
        #pragma unroll
        for (int ks = 0; ks < 8; ++ks) {
            Frag16 A, Bf;
            #pragma unroll
            for (int v = 0; v < 8; ++v) {
                int ka = ks * 32 + ((v & 3) * 2) + ((v >> 2) * 16) + half * 8;
                A.u[v]  = *(const unsigned*)&Qi[m * 256 + ka];
                int kb = ks * 32 + ((v & 3) * 2) + ((v >> 2) * 16) + half * 8;
                Bf.u[v] = *(const unsigned*)&Kh[(wv * 16 + nn) * 256 + kb];
            }
            acc = __builtin_amdgcn_wmma_f32_16x16x32_f16(
                false, A.v, false, Bf.v, (short)0, acc, false, false);
        }
        #pragma unroll
        for (int r = 0; r < 8; ++r) {
            int mi = r + half * 8;
            int i  = ib * 16 + mi;
            int j  = wv * 16 + nn;
            float p = pos[(size_t)(nd * 256 + i) * 256 + j];
            Srow[mi * 256 + j] = (acc[r] + p) * 0.25f;
        }
        __syncthreads();

        // softmax: wave wv owns row wv (8 elements/lane, shfl reductions);
        // normalized values written transposed into Tb[j][i_local]
        {
            float vbuf[8];
            float mx = -3.4e38f;
            #pragma unroll
            for (int u = 0; u < 8; ++u) {
                vbuf[u] = Srow[wv * 256 + lane * 8 + u];
                mx = fmaxf(mx, vbuf[u]);
            }
            #pragma unroll
            for (int off = 16; off >= 1; off >>= 1)
                mx = fmaxf(mx, __shfl_xor(mx, off, 32));
            float sum = 0.f;
            #pragma unroll
            for (int u = 0; u < 8; ++u) { vbuf[u] = __expf(vbuf[u] - mx); sum += vbuf[u]; }
            #pragma unroll
            for (int off = 16; off >= 1; off >>= 1)
                sum += __shfl_xor(sum, off, 32);
            float inv = 1.0f / sum;
            #pragma unroll
            for (int u = 0; u < 8; ++u) {
                int j = lane * 8 + u;
                Tb[j * 16 + wv] = (_Float16)(vbuf[u] * inv);
            }
        }
        __syncthreads();
        // coalesced P^T store: thread t -> row j = t>>1, 16B piece = t&1
        {
            int j = tid >> 1, part = tid & 1;
            *(uint4*)(Pt + sliceBase + (size_t)j * 256 + ib * 16 + part * 8) =
                ((const uint4*)Tb)[tid];
        }
    }
}

// ---------------------------------------------------------------------------
// Kernel 3: net = V @ P per slice.  M = h, K = i, N = j.
// P^T slice async-loaded into LDS; V A-fragments hoisted into registers.
// out index collapses to bnd*65536 + h*256 + j (channel = n*16+d).
// ---------------------------------------------------------------------------
__global__ __launch_bounds__(512) void attn_out_kernel(
    const _Float16* __restrict__ vM, const _Float16* __restrict__ Pt,
    float* __restrict__ out)
{
    __shared__ __align__(16) _Float16 Ph[65536];  // P^T [j][i]  128 KB

    const int tid = threadIdx.x;
    const int bnd = blockIdx.x;
    const size_t sliceBase = (size_t)bnd * 65536;

    {
        const unsigned phBase = lds_addr_of(Ph);
        const _Float16* src = Pt + sliceBase;
        for (int i = tid; i < 8192; i += 512)
            async_copy_b128(phBase + i * 16, src + (size_t)i * 8);
    }

    const int lane = tid & 31, wv = tid >> 5;     // wave = h-tile
    const int m = lane & 15, half = lane >> 4, nn = m;

    Frag16 Afr[8];                                 // V rows, hoisted (64 VGPRs)
    #pragma unroll
    for (int ks = 0; ks < 8; ++ks) {
        #pragma unroll
        for (int v = 0; v < 8; ++v) {
            int ka = ks * 32 + ((v & 3) * 2) + ((v >> 2) * 16) + half * 8;
            Afr[ks].u[v] =
                *(const unsigned*)&vM[sliceBase + (size_t)(wv * 16 + m) * 256 + ka];
        }
    }
    wait_async0();
    __syncthreads();

    for (int jt = 0; jt < 16; ++jt) {
        v8f acc = {};
        #pragma unroll
        for (int ks = 0; ks < 8; ++ks) {
            Frag16 Bf;
            #pragma unroll
            for (int v = 0; v < 8; ++v) {
                int kb = ks * 32 + ((v & 3) * 2) + ((v >> 2) * 16) + half * 8;
                Bf.u[v] = *(const unsigned*)&Ph[(jt * 16 + nn) * 256 + kb];
            }
            acc = __builtin_amdgcn_wmma_f32_16x16x32_f16(
                false, Afr[ks].v, false, Bf.v, (short)0, acc, false, false);
        }
        #pragma unroll
        for (int r = 0; r < 8; ++r) {
            int h = wv * 16 + r + half * 8;
            int j = jt * 16 + nn;
            out[sliceBase + (size_t)h * 256 + j] = acc[r];
        }
    }
}

// ---------------------------------------------------------------------------
extern "C" void kernel_launch(void* const* d_in, const int* in_sizes, int n_in,
                              void* d_out, int out_size, void* d_ws, size_t ws_size,
                              hipStream_t stream) {
    const float* x    = (const float*)d_in[0];   // (8,64,256,256)
    const float* Wq   = (const float*)d_in[1];   // (192,64)
    const float* bq   = (const float*)d_in[2];   // (192,)
    const float* pos  = (const float*)d_in[3];   // (4,16,256,256)
    float* out        = (float*)d_out;           // (8,64,256,256)

    _Float16* wsh = (_Float16*)d_ws;
    _Float16* qT  = wsh;                         // [512][256][256] f16 (w-major)
    _Float16* kT  = wsh + (size_t)33554432;      // [512][256][256] f16 (w-major)
    _Float16* vM  = wsh + (size_t)67108864;      // [512][256][256] f16 (h-major)
    _Float16* Pt  = wsh + (size_t)100663296;     // [512][256][256] f16 (j-major)

    qkv_proj_kernel<<<dim3(512, 8), 256, 0, stream>>>(x, Wq, bq, qT, kT, vM);
    attn_scores_kernel<<<512, 512, 0, stream>>>(qT, kT, pos, Pt);
    attn_out_kernel<<<512, 512, 0, stream>>>(vM, Pt, out);
}
